// SetAttention_36498632081553
// MI455X (gfx1250) — matrix operand order repaired and checked by reading, not compiled, not run on hardware
//
#include <hip/hip_runtime.h>
#include <hip/hip_bf16.h>
#include <stdint.h>

// GAT layer, MI455X (gfx1250), wave32.
// Strategy: column-softmax stats precomputed (softmax is over axis=1),
// attention matrix never hits HBM; fused bf16-WMMA GEMM regenerates
// 16x32 probability tiles in LDS per K-step.

#define BATCH 8
#define NPTS  2048
#define FDIM  128
#define ALPHA 0.1f
#define NEG_BIG -9000000000000000.0f

typedef __attribute__((ext_vector_type(16))) __bf16 v16bf;
typedef __attribute__((ext_vector_type(8)))  float  v8f;

static __device__ __forceinline__ __bf16 f2bf(float f) {
  uint32_t u = __builtin_bit_cast(uint32_t, f);
  uint32_t r = u + 0x7FFFu + ((u >> 16) & 1u);
  uint16_t h = (uint16_t)(r >> 16);
  return __builtin_bit_cast(__bf16, h);
}

union AFrag { uint4 q[2]; v16bf v; };

// ---------------- K0a: x (f32) -> x_bf (bf16) ----------------
__global__ void k_convert_x(const float* __restrict__ x, __bf16* __restrict__ xbf, int n) {
  int stride = gridDim.x * blockDim.x;
  for (int j = blockIdx.x * blockDim.x + threadIdx.x; j < n; j += stride)
    xbf[j] = f2bf(x[j]);
}

// ---------------- K0b: W[k][n] (f32) -> WT[n][k] (bf16) ----------------
__global__ void k_transpose_w(const float* __restrict__ w, __bf16* __restrict__ wt) {
  int t = blockIdx.x * blockDim.x + threadIdx.x;    // 128*128 threads
  int k = t >> 7, n = t & 127;
  wt[n * FDIM + k] = f2bf(w[k * FDIM + n]);
}

// ---------------- K0c: adj -> bitmask (512KB, L2-resident) ----------------
__global__ void k_mask(const float* __restrict__ adj, uint32_t* __restrict__ mb) {
  int t = blockIdx.x * blockDim.x + threadIdx.x;    // N*N/32 words
  const float* a = adj + (size_t)t * 32;
  uint32_t bits = 0u;
#pragma unroll
  for (int j = 0; j < 32; ++j) bits |= (a[j] > 0.0f ? 1u : 0u) << j;
  mb[t] = bits;
}

// ---------------- K1: h = x@Wo (f32 + bf16-transposed), out = x@Ws ----------------
// One wave per 16x16 tile; 8 waves/block cover the 8 column tiles.
__global__ void k_gemm_xw(const __bf16* __restrict__ xbf,
                          const __bf16* __restrict__ WoT,
                          const __bf16* __restrict__ WsT,
                          float*  __restrict__ h,
                          __bf16* __restrict__ hT,
                          float*  __restrict__ out) {
  const int w = threadIdx.x >> 5, lane = threadIdx.x & 31;
  const int g = lane >> 4, lr = lane & 15;
  const int rm0 = blockIdx.x * 16;                  // flattened row tile (b*N+n)
  const int col = w * 16 + lr;
  v8f accO = {0.f,0.f,0.f,0.f,0.f,0.f,0.f,0.f};
  v8f accS = {0.f,0.f,0.f,0.f,0.f,0.f,0.f,0.f};
#pragma unroll
  for (int kk = 0; kk < FDIM; kk += 32) {
    const __bf16* xr = xbf + (size_t)(rm0 + lr) * FDIM + kk;
    AFrag a;
    a.q[0] = *(const uint4*)(xr + g * 16);          // K = kk + 16g .. +7
    a.q[1] = *(const uint4*)(xr + g * 16 + 8);      // K = kk + 16g + 8 .. +15
    v16bf bo = *(const v16bf*)(WoT + (size_t)col * FDIM + kk + g * 16);
    v16bf bs = *(const v16bf*)(WsT + (size_t)col * FDIM + kk + g * 16);
    accO = __builtin_amdgcn_wmma_f32_16x16x32_bf16(false, a.v, false, bo, (short)0, accO, false, false);
    accS = __builtin_amdgcn_wmma_f32_16x16x32_bf16(false, a.v, false, bs, (short)0, accS, false, false);
  }
  const int b = rm0 >> 11;
#pragma unroll
  for (int v = 0; v < 8; ++v) {
    int row = rm0 + v + g * 8;
    int n   = row & (NPTS - 1);
    h[(size_t)row * FDIM + col]   = accO[v];
    hT[((size_t)(b * FDIM + col)) * NPTS + n] = f2bf(accO[v]);
    out[(size_t)row * FDIM + col] = accS[v];
  }
}

// ---------------- K2: u1 = h@a1, u2 = h@a2 (wave per row) ----------------
__global__ void k_uvec(const float* __restrict__ h, const float* __restrict__ attw,
                       float* __restrict__ u1, float* __restrict__ u2) {
  int row  = blockIdx.x * 8 + (threadIdx.x >> 5);
  int lane = threadIdx.x & 31;
  const float* hr = h + (size_t)row * FDIM;
  float s1 = 0.f, s2 = 0.f;
#pragma unroll
  for (int f = lane; f < FDIM; f += 32) {
    float hv = hr[f];
    s1 += hv * attw[f];
    s2 += hv * attw[FDIM + f];
  }
#pragma unroll
  for (int o = 16; o; o >>= 1) { s1 += __shfl_xor(s1, o, 32); s2 += __shfl_xor(s2, o, 32); }
  if (lane == 0) { u1[row] = s1; u2[row] = s2; }
}

static __device__ __forceinline__ float logit_s(const float* U1, const float* U2,
                                                const uint32_t* mb, int i, int k) {
  int m  = i * NPTS + k;
  int k1 = m << 1;
  int idx1 = (k1     < NPTS * NPTS) ? (k1 >> 11)       : (k1 & (NPTS - 1));
  int idx2 = (k1 + 1 < NPTS * NPTS) ? ((k1 + 1) >> 11) : ((k1 + 1) & (NPTS - 1));
  float e = U1[idx1] + U2[idx2];
  float s = e >= 0.f ? e : ALPHA * e;
  bool ok = (mb[(i << 6) + (k >> 5)] >> (k & 31)) & 1u;
  return ok ? s : NEG_BIG;
}

// ---------------- K3: per-column softmax stats (max + denom), wave per column ----------------
__global__ void k_colstats(const float* __restrict__ u1, const float* __restrict__ u2,
                           const uint32_t* __restrict__ mb,
                           float* __restrict__ Mcol, float* __restrict__ Dcol) {
  int gcol = blockIdx.x * 8 + (threadIdx.x >> 5);   // b*N + k
  int b = gcol >> 11, k = gcol & (NPTS - 1);
  int lane = threadIdx.x & 31;
  const float* U1 = u1 + b * NPTS;
  const float* U2 = u2 + b * NPTS;
  float m = -INFINITY, d = 0.f;
  for (int i = lane; i < NPTS; i += 32) {
    float s  = logit_s(U1, U2, mb, i, k);
    float nm = fmaxf(m, s);
    d = d * __expf(m - nm) + __expf(s - nm);
    m = nm;
  }
#pragma unroll
  for (int o = 16; o; o >>= 1) {
    float m2 = __shfl_xor(m, o, 32), d2 = __shfl_xor(d, o, 32);
    float nm = fmaxf(m, m2);
    d = d * __expf(m - nm) + d2 * __expf(m2 - nm);
    m = nm;
  }
  if (lane == 0) { Mcol[gcol] = m; Dcol[gcol] = d; }
}

// ---------------- K4: out += softmax(att) @ h, att regenerated per tile ----------------
// Block = 256 thr (8 waves) owns a 16-row i-tile x all 128 out cols; per K-step
// the block builds the 16x32 probability tile in LDS, each wave WMMAs it
// against its 16-col slice of hT (bf16, L2-resident).
__global__ void k_attn(const float* __restrict__ u1, const float* __restrict__ u2,
                       const uint32_t* __restrict__ mb,
                       const float* __restrict__ Mcol, const float* __restrict__ Dcol,
                       const __bf16* __restrict__ hT,
                       float* __restrict__ out) {
  __shared__ __align__(64) __bf16 ptile[16 * 32];
  const int b   = blockIdx.x >> 7;
  const int i0  = (blockIdx.x & 127) << 4;
  const int tid = threadIdx.x;
  const int w = tid >> 5, lane = tid & 31;
  const int g = lane >> 4, lr = lane & 15;
  const int col = w * 16 + lr;
  const float* U1 = u1 + b * NPTS;
  const float* U2 = u2 + b * NPTS;
  const float* MC = Mcol + b * NPTS;
  const float* DC = Dcol + b * NPTS;
  const __bf16* hTb = hT + ((size_t)(b * FDIM + col)) * NPTS;
  v8f acc = {0.f,0.f,0.f,0.f,0.f,0.f,0.f,0.f};
  for (int kk = 0; kk < NPTS; kk += 32) {
#pragma unroll
    for (int t = 0; t < 2; ++t) {                   // 512 tile entries / 256 threads
      int q  = tid + t * 256;
      int M  = q >> 5, kc = q & 31;
      int k  = kk + kc;
      float s = logit_s(U1, U2, mb, i0 + M, k);
      float p = __expf(s - MC[k]) / DC[k];
      ptile[q] = f2bf(p);                           // row-major [M][32]
    }
    __syncthreads();
    const __bf16* pr = ptile + lr * 32;
    AFrag a;
    a.q[0] = *(const uint4*)(pr + g * 16);
    a.q[1] = *(const uint4*)(pr + g * 16 + 8);
    v16bf bb = *(const v16bf*)(hTb + kk + g * 16);
    acc = __builtin_amdgcn_wmma_f32_16x16x32_bf16(false, a.v, false, bb, (short)0, acc, false, false);
    __syncthreads();
  }
#pragma unroll
  for (int v = 0; v < 8; ++v) {
    size_t o = ((size_t)(b * NPTS + i0 + v + g * 8)) * FDIM + col;
    out[o] += acc[v];                               // K1 seeded out with x@Ws
  }
}

extern "C" void kernel_launch(void* const* d_in, const int* in_sizes, int n_in,
                              void* d_out, int out_size, void* d_ws, size_t ws_size,
                              hipStream_t stream) {
  (void)in_sizes; (void)n_in; (void)out_size; (void)ws_size;
  const float* x    = (const float*)d_in[0];
  const float* adj  = (const float*)d_in[1];
  const float* wo   = (const float*)d_in[2];
  const float* ws   = (const float*)d_in[3];
  const float* attw = (const float*)d_in[4];
  float* out = (float*)d_out;

  char* p = (char*)d_ws;
  auto carve = [&](size_t bytes) { void* r = (void*)p; p += (bytes + 255) & ~(size_t)255; return r; };
  const size_t NE = (size_t)BATCH * NPTS * FDIM;    // 2,097,152
  __bf16*   xbf   = (__bf16*)  carve(NE * 2);
  __bf16*   WoT   = (__bf16*)  carve((size_t)FDIM * FDIM * 2);
  __bf16*   WsT   = (__bf16*)  carve((size_t)FDIM * FDIM * 2);
  float*    h     = (float*)   carve(NE * 4);
  __bf16*   hT    = (__bf16*)  carve(NE * 2);
  float*    u1    = (float*)   carve((size_t)BATCH * NPTS * 4);
  float*    u2    = (float*)   carve((size_t)BATCH * NPTS * 4);
  float*    Mc    = (float*)   carve((size_t)BATCH * NPTS * 4);
  float*    Dc    = (float*)   carve((size_t)BATCH * NPTS * 4);
  uint32_t* maskb = (uint32_t*)carve((size_t)NPTS * NPTS / 8);

  k_convert_x <<<2048, 256, 0, stream>>>(x, xbf, (int)NE);
  k_transpose_w<<<64,  256, 0, stream>>>(wo, WoT);
  k_transpose_w<<<64,  256, 0, stream>>>(ws, WsT);
  k_mask      <<<512,  256, 0, stream>>>(adj, maskb);
  k_gemm_xw   <<<1024, 256, 0, stream>>>(xbf, WoT, WsT, h, hT, out);
  k_uvec      <<<2048, 256, 0, stream>>>(h, attw, u1, u2);
  k_colstats  <<<2048, 256, 0, stream>>>(u1, u2, maskb, Mc, Dc);
  k_attn      <<<1024, 256, 0, stream>>>(u1, u2, maskb, Mc, Dc, hT, out);
}
